// ClassificationModel_9156870275081
// MI455X (gfx1250) — matrix operand order repaired and checked by reading, not compiled, not run on hardware
//
#include <hip/hip_runtime.h>
#include <hip/hip_bf16.h>

// ---------------------------------------------------------------------------
// CDNA5 (gfx1250) implementation of the CNN + transformer classifier.
// Big GEMMs: v_wmma_f32_16x16x32_f16 with double-buffered async global->LDS
// operand staging (global_load_async_to_lds_b128 / s_wait_asynccnt).
// ---------------------------------------------------------------------------

typedef __attribute__((ext_vector_type(16))) _Float16 v16h;
typedef __attribute__((ext_vector_type(8)))  _Float16 v8h;
typedef __attribute__((ext_vector_type(8)))  float    v8f;

#define NB   64
#define LW   128
#define WW   256
#define DM   384
#define HH   8
#define EE   48
#define DFFN 1536
#define NLAY 4
#define LIN  2048
#define ROWS (NB * LW)   // 8192 token rows

// 16 bytes of global memory -> LDS, asynchronously (ASYNCcnt-tracked).
__device__ __forceinline__ void async_copy_b128(unsigned lds_addr,
                                                const void* gaddr) {
  asm volatile("global_load_async_to_lds_b128 %0, %1, off"
               :: "v"(lds_addr), "v"((unsigned long long)(uintptr_t)gaddr)
               : "memory");
}
__device__ __forceinline__ void wait_async0() {
  asm volatile("s_wait_asynccnt 0x0" ::: "memory");
}
__device__ __forceinline__ unsigned lds_addr32(const void* p) {
  // Generic LDS address: low 32 bits are the LDS byte offset.
  return (unsigned)(uintptr_t)p;
}

// ------------------------------- f32 -> f16 --------------------------------
__global__ void f32_to_f16_kernel(const float* __restrict__ src,
                                  _Float16* __restrict__ dst, long n) {
  long i = (long)blockIdx.x * blockDim.x + threadIdx.x;
  if (i < n) dst[i] = (_Float16)src[i];
}

// --------------------------- CNN window encoder ----------------------------
// One block per window: conv(7,pad3)+relu+maxpool2  x3, channels 1->4->16->64.
__global__ __launch_bounds__(256) void conv_encoder_kernel(
    const float* __restrict__ x,   // (ROWS, 256)
    const float* __restrict__ w0, const float* __restrict__ b0,   // (4,1,7)
    const float* __restrict__ w1, const float* __restrict__ b1,   // (16,4,7)
    const float* __restrict__ w2, const float* __restrict__ b2,   // (64,16,7)
    _Float16* __restrict__ feat)   // (ROWS, 2048)
{
  __shared__ float s_in[256];
  __shared__ float s_a[4 * 256];
  __shared__ float s_p0[4 * 128];
  __shared__ float s_b[16 * 128];
  __shared__ float s_p1[16 * 64];
  __shared__ float s_c[64 * 64];

  const int win = blockIdx.x;
  const int tid = threadIdx.x;

  s_in[tid] = x[(long)win * 256 + tid];
  __syncthreads();

  #pragma unroll
  for (int co = 0; co < 4; ++co) {
    float acc = b0[co];
    #pragma unroll
    for (int t = 0; t < 7; ++t) {
      int p = tid + t - 3;
      float xv = (p >= 0 && p < 256) ? s_in[p] : 0.f;
      acc += w0[co * 7 + t] * xv;
    }
    s_a[co * 256 + tid] = acc > 0.f ? acc : 0.f;
  }
  __syncthreads();
  for (int i = tid; i < 4 * 128; i += 256) {
    int c = i >> 7, p = i & 127;
    float u = s_a[c * 256 + 2 * p], v = s_a[c * 256 + 2 * p + 1];
    s_p0[i] = u > v ? u : v;
  }
  __syncthreads();

  for (int i = tid; i < 16 * 128; i += 256) {
    int co = i >> 7, p = i & 127;
    float acc = b1[co];
    for (int ci = 0; ci < 4; ++ci) {
      const float* wr = w1 + (co * 4 + ci) * 7;
      #pragma unroll
      for (int t = 0; t < 7; ++t) {
        int pp = p + t - 3;
        float xv = (pp >= 0 && pp < 128) ? s_p0[ci * 128 + pp] : 0.f;
        acc += wr[t] * xv;
      }
    }
    s_b[i] = acc > 0.f ? acc : 0.f;
  }
  __syncthreads();
  for (int i = tid; i < 16 * 64; i += 256) {
    int c = i >> 6, p = i & 63;
    float u = s_b[c * 128 + 2 * p], v = s_b[c * 128 + 2 * p + 1];
    s_p1[i] = u > v ? u : v;
  }
  __syncthreads();

  for (int i = tid; i < 64 * 64; i += 256) {
    int co = i >> 6, p = i & 63;
    float acc = b2[co];
    for (int ci = 0; ci < 16; ++ci) {
      const float* wr = w2 + (co * 16 + ci) * 7;
      #pragma unroll
      for (int t = 0; t < 7; ++t) {
        int pp = p + t - 3;
        float xv = (pp >= 0 && pp < 64) ? s_p1[ci * 64 + pp] : 0.f;
        acc += wr[t] * xv;
      }
    }
    s_c[i] = acc > 0.f ? acc : 0.f;
  }
  __syncthreads();

  _Float16* out = feat + (long)win * 2048;
  for (int i = tid; i < 64 * 32; i += 256) {
    int c = i >> 5, p = i & 31;
    float u = s_c[c * 64 + 2 * p], v = s_c[c * 64 + 2 * p + 1];
    out[i] = (_Float16)(u > v ? u : v);
  }
}

// ------------------------------- WMMA GEMM ---------------------------------
// C(MxN) = [relu](A16(MxK,row-major) @ B16(KxN,row-major) + bias)
// Block = 256 threads = 8 waves arranged 2(M) x 4(N); block tile 32x64,
// one 16x16 C tile per wave. K stepped by 32. Operand tiles staged in LDS
// via double-buffered global_load_async_to_lds_b128 (ASYNCcnt).
// Requires M % 32 == 0, N % 64 == 0, K % 32 == 0 (true for all call sites).
__global__ __launch_bounds__(256) void wmma_gemm_kernel(
    const _Float16* __restrict__ A, const _Float16* __restrict__ B,
    const float* __restrict__ bias, float* __restrict__ C32,
    _Float16* __restrict__ C16, int M, int N, int K, int relu)
{
  __shared__ _Float16 sA[2][32 * 32];   // [buf][m_local][k_local]   2 KB each
  __shared__ _Float16 sB[2][32 * 64];   // [buf][k_local][n_local]   4 KB each

  const int tid  = threadIdx.x;
  const int lane = tid & 31;
  const int wave = tid >> 5;
  const int wm = wave >> 2, wn = wave & 3;
  const int m0 = blockIdx.x * 32;
  const int n0 = blockIdx.y * 64;

  // ---- staging coordinates ----
  // B tile: 32 rows x 128B; 256 lanes x 16B  (row = tid/8, chunk = tid%8)
  const int brow = tid >> 3, bch = tid & 7;
  // A tile: 32 rows x  64B; 128 lanes x 16B  (waves 0-3 only: uniform branch)
  const int arow = (tid & 127) >> 2, ach = tid & 3;
  const bool doA = (tid < 128);

  const _Float16* gB = B + (long)brow * N + n0 + bch * 8;
  const _Float16* gA = A + (long)(m0 + arow) * K + ach * 8;
  const unsigned lB0 = lds_addr32(&sB[0][brow * 64 + bch * 8]);
  const unsigned lB1 = lds_addr32(&sB[1][brow * 64 + bch * 8]);
  const unsigned lA0 = lds_addr32(&sA[0][arow * 32 + ach * 8]);
  const unsigned lA1 = lds_addr32(&sA[1][arow * 32 + ach * 8]);

  const int nsteps = K >> 5;

  // prologue: stage step 0 into buffer 0
  async_copy_b128(lB0, gB);
  if (doA) async_copy_b128(lA0, gA);
  wait_async0();
  __syncthreads();

  // ---- fragment coordinates (CDNA5 ISA 7.12.2 striping) ----
  const int fm    = (wm << 4) + (lane & 15);          // A row in block tile
  const int khalf = (lane >> 4) << 3;                 // lanes 16-31: K +8
  const int frB   = (lane & 15) + ((lane >> 4) << 4); // B row in K tile
  const int fnc   = wn << 4;                          // B col base

  v8f acc = {};
  for (int s = 0; s < nsteps; ++s) {
    const int cur = s & 1;
    if (s + 1 < nsteps) {   // issue next tile's async copies (other buffer)
      const _Float16* gBn = gB + (long)(s + 1) * 32 * N;
      const _Float16* gAn = gA + (long)(s + 1) * 32;
      async_copy_b128(cur ? lB0 : lB1, gBn);
      if (doA) async_copy_b128(cur ? lA0 : lA1, gAn);
    }
    const _Float16* sa = cur ? &sA[1][0] : &sA[0][0];
    const _Float16* sb = cur ? &sB[1][0] : &sB[0][0];
    v16h a, b;
    *(v8h*)&a       = *(const v8h*)(sa + fm * 32 + khalf);        // K 0..7
    *((v8h*)&a + 1) = *(const v8h*)(sa + fm * 32 + khalf + 16);   // K 16..23
    for (int i = 0; i < 16; ++i)
      b[i] = sb[(khalf + (i & 7) + ((i >> 3) << 4)) * 64 + fnc + (lane & 15)];
    acc = __builtin_amdgcn_wmma_f32_16x16x32_f16(
        /*neg_a=*/false, a, /*neg_b=*/false, b,
        /*c_mod=*/(short)0, acc, /*reuse_a=*/false, /*reuse_b=*/false);
    wait_async0();        // next buffer's copy complete (this wave)
    __syncthreads();      // ... and visible to / done by all waves
  }

  // D layout: VGPR r, lanes 0-15 -> (M=base+r, N=n+lane); lanes 16-31 -> M+8.
  const int crow = m0 + (wm << 4) + ((lane >> 4) << 3);
  const int ccol = n0 + fnc + (lane & 15);
  #pragma unroll
  for (int r = 0; r < 8; ++r) {
    float v = acc[r];
    if (bias) v += bias[ccol];
    if (relu) v = v > 0.f ? v : 0.f;
    long idx = (long)(crow + r) * N + ccol;
    if (C32) C32[idx] = v;
    if (C16) C16[idx] = (_Float16)v;
  }
}

// --------------------------- positional encoding ---------------------------
__global__ void add_pe_kernel(float* __restrict__ t32,
                              _Float16* __restrict__ t16, long total) {
  long idx = (long)blockIdx.x * blockDim.x + threadIdx.x;
  if (idx >= total) return;
  int d = (int)(idx % DM);
  int l = (int)((idx / DM) % LW);
  float ex  = (float)(2 * (d / 2)) / (float)DM;
  float ang = (float)l * powf(10000.f, -ex);
  float pe  = (d & 1) ? cosf(ang) : sinf(ang);
  float v = t32[idx] + pe;
  t32[idx] = v;
  t16[idx] = (_Float16)v;
}

// ------------------------------- attention ---------------------------------
// One block per (n, h). K/V staged in LDS; one thread per query position.
__global__ __launch_bounds__(128) void attention_kernel(
    const float* __restrict__ q, const float* __restrict__ k,
    const float* __restrict__ v, _Float16* __restrict__ o16)
{
  const int n = blockIdx.x / HH;
  const int h = blockIdx.x % HH;
  const int l = threadIdx.x;

  __shared__ float sk[LW * EE];
  __shared__ float sv[LW * EE];
  for (int i = threadIdx.x; i < LW * EE; i += 128) {
    int s = i / EE, e = i % EE;
    long off = ((long)n * LW + s) * DM + h * EE + e;
    sk[i] = k[off];
    sv[i] = v[off];
  }
  __syncthreads();

  const long qoff = ((long)n * LW + l) * DM + h * EE;
  float qe[EE];
  #pragma unroll
  for (int e = 0; e < EE; ++e) qe[e] = q[qoff + e];

  const float temp = rsqrtf((float)EE);
  float scores[LW];
  float mx = -1e30f;
  for (int s = 0; s < LW; ++s) {
    float acc = 0.f;
    #pragma unroll 8
    for (int e = 0; e < EE; ++e) acc += qe[e] * sk[s * EE + e];
    acc *= temp;
    scores[s] = acc;
    mx = fmaxf(mx, acc);
  }
  float sum = 0.f;
  for (int s = 0; s < LW; ++s) {
    float p = __expf(scores[s] - mx);
    scores[s] = p;
    sum += p;
  }
  const float inv = 1.f / sum;
  for (int e = 0; e < EE; ++e) {
    float acc = 0.f;
    #pragma unroll 8
    for (int s = 0; s < LW; ++s) acc += scores[s] * sv[s * EE + e];
    o16[qoff + e] = (_Float16)(acc * inv);
  }
}

// --------------------------- residual + LayerNorm --------------------------
__global__ __launch_bounds__(128) void add_ln_kernel(
    float* __restrict__ t32, const float* __restrict__ y,
    const float* __restrict__ g, const float* __restrict__ be,
    _Float16* __restrict__ t16)
{
  const int row = blockIdx.x;
  const int tid = threadIdx.x;
  __shared__ float rbuf[128];

  float* tr = t32 + (long)row * DM;
  const float* yr = y + (long)row * DM;

  float v0 = tr[tid]       + yr[tid];
  float v1 = tr[tid + 128] + yr[tid + 128];
  float v2 = tr[tid + 256] + yr[tid + 256];

  rbuf[tid] = v0 + v1 + v2;
  __syncthreads();
  for (int s = 64; s > 0; s >>= 1) {
    if (tid < s) rbuf[tid] += rbuf[tid + s];
    __syncthreads();
  }
  const float mean = rbuf[0] / (float)DM;
  __syncthreads();

  float d0 = v0 - mean, d1 = v1 - mean, d2 = v2 - mean;
  rbuf[tid] = d0 * d0 + d1 * d1 + d2 * d2;
  __syncthreads();
  for (int s = 64; s > 0; s >>= 1) {
    if (tid < s) rbuf[tid] += rbuf[tid + s];
    __syncthreads();
  }
  const float rinv = rsqrtf(rbuf[0] / (float)DM + 1e-5f);

  #pragma unroll
  for (int j = 0; j < 3; ++j) {
    int d = tid + 128 * j;
    float dv = (j == 0 ? d0 : (j == 1 ? d1 : d2));
    float o = dv * rinv * g[d] + be[d];
    tr[d] = o;
    t16[(long)row * DM + d] = (_Float16)o;
  }
}

// ------------------------------- classifier --------------------------------
__global__ __launch_bounds__(256) void classifier_kernel(
    const float* __restrict__ t32, const float* __restrict__ cls_w,
    const float* __restrict__ cls_b, float* __restrict__ out)
{
  const int n = blockIdx.x;
  float acc = 0.f;
  for (int i = threadIdx.x; i < LW * DM; i += 256) {
    acc += t32[(long)n * LW * DM + i] * cls_w[i % DM];
  }
  __shared__ float rbuf[256];
  rbuf[threadIdx.x] = acc;
  __syncthreads();
  for (int s = 128; s > 0; s >>= 1) {
    if (threadIdx.x < s) rbuf[threadIdx.x] += rbuf[threadIdx.x + s];
    __syncthreads();
  }
  if (threadIdx.x == 0) out[n] = rbuf[0] / (float)LW + cls_b[0];
}

// ---------------------------------------------------------------------------
extern "C" void kernel_launch(void* const* d_in, const int* in_sizes, int n_in,
                              void* d_out, int out_size, void* d_ws, size_t ws_size,
                              hipStream_t stream) {
  const float* x       = (const float*)d_in[0];
  const float* conv_w0 = (const float*)d_in[1];
  const float* conv_b0 = (const float*)d_in[2];
  const float* conv_w1 = (const float*)d_in[3];
  const float* conv_b1 = (const float*)d_in[4];
  const float* conv_w2 = (const float*)d_in[5];
  const float* conv_b2 = (const float*)d_in[6];
  const float* embed_w = (const float*)d_in[7];
  const float* embed_b = (const float*)d_in[8];
  const float* Wq = (const float*)d_in[9];
  const float* Wk = (const float*)d_in[10];
  const float* Wv = (const float*)d_in[11];
  const float* Wo = (const float*)d_in[12];
  const float* W1 = (const float*)d_in[13];
  const float* W2 = (const float*)d_in[14];
  const float* bq = (const float*)d_in[15];
  const float* bk = (const float*)d_in[16];
  const float* bv = (const float*)d_in[17];
  const float* bo = (const float*)d_in[18];
  const float* b1 = (const float*)d_in[19];
  const float* b2 = (const float*)d_in[20];
  const float* g1  = (const float*)d_in[21];
  const float* be1 = (const float*)d_in[22];
  const float* g2  = (const float*)d_in[23];
  const float* be2 = (const float*)d_in[24];
  const float* cls_w = (const float*)d_in[25];
  const float* cls_b = (const float*)d_in[26];
  float* out = (float*)d_out;

  // ---------------- workspace carve-up (256B aligned slices) ---------------
  char* ws = (char*)d_ws;
  size_t off = 0;
  auto carve = [&](size_t bytes) -> void* {
    off = (off + 255) & ~(size_t)255;
    void* p = ws + off;
    off += bytes;
    return p;
  };

  const long nEmbW = (long)LIN * DM;
  const long nProj = (long)NLAY * DM * DM;
  const long nFF   = (long)NLAY * DM * DFFN;

  _Float16* feat16 = (_Float16*)carve((long)ROWS * LIN * 2);
  _Float16* eW16   = (_Float16*)carve(nEmbW * 2);
  _Float16* Wq16   = (_Float16*)carve(nProj * 2);
  _Float16* Wk16   = (_Float16*)carve(nProj * 2);
  _Float16* Wv16   = (_Float16*)carve(nProj * 2);
  _Float16* Wo16   = (_Float16*)carve(nProj * 2);
  _Float16* W116   = (_Float16*)carve(nFF * 2);
  _Float16* W216   = (_Float16*)carve(nFF * 2);
  float*    t32    = (float*)carve((long)ROWS * DM * 4);
  _Float16* t16    = (_Float16*)carve((long)ROWS * DM * 2);
  float*    q32    = (float*)carve((long)ROWS * DM * 4);
  float*    k32    = (float*)carve((long)ROWS * DM * 4);
  float*    v32    = (float*)carve((long)ROWS * DM * 4);
  _Float16* o16    = (_Float16*)carve((long)ROWS * DM * 2);
  float*    y32    = (float*)carve((long)ROWS * DM * 4);
  _Float16* mid16  = (_Float16*)carve((long)ROWS * DFFN * 2);
  (void)ws_size; (void)n_in; (void)in_sizes; (void)out_size;

  auto cvt = [&](const float* src, _Float16* dst, long n) {
    f32_to_f16_kernel<<<dim3((unsigned)((n + 255) / 256)), dim3(256), 0, stream>>>(src, dst, n);
  };

  // 1) weights -> f16
  cvt(embed_w, eW16, nEmbW);
  cvt(Wq, Wq16, nProj);  cvt(Wk, Wk16, nProj);
  cvt(Wv, Wv16, nProj);  cvt(Wo, Wo16, nProj);
  cvt(W1, W116, nFF);    cvt(W2, W216, nFF);

  // 2) CNN window encoder -> feat16 (ROWS x 2048)
  conv_encoder_kernel<<<dim3(ROWS), dim3(256), 0, stream>>>(
      x, conv_w0, conv_b0, conv_w1, conv_b1, conv_w2, conv_b2, feat16);

  // 3) embed GEMM: t32 = relu(feat16 @ eW16 + embed_b)
  wmma_gemm_kernel<<<dim3(ROWS / 32, DM / 64), dim3(256), 0, stream>>>(
      feat16, eW16, embed_b, t32, (_Float16*)nullptr, ROWS, DM, LIN, 1);

  // 4) + positional encoding (also emits t16)
  {
    long total = (long)ROWS * DM;
    add_pe_kernel<<<dim3((unsigned)((total + 255) / 256)), dim3(256), 0, stream>>>(
        t32, t16, total);
  }

  // 5) transformer layers
  for (int l = 0; l < NLAY; ++l) {
    const _Float16* wq = Wq16 + (long)l * DM * DM;
    const _Float16* wk = Wk16 + (long)l * DM * DM;
    const _Float16* wv = Wv16 + (long)l * DM * DM;
    const _Float16* wo = Wo16 + (long)l * DM * DM;
    const _Float16* w1 = W116 + (long)l * DM * DFFN;
    const _Float16* w2 = W216 + (long)l * DFFN * DM;

    dim3 gProj(ROWS / 32, DM / 64);
    wmma_gemm_kernel<<<gProj, dim3(256), 0, stream>>>(
        t16, wq, bq + (long)l * DM, q32, (_Float16*)nullptr, ROWS, DM, DM, 0);
    wmma_gemm_kernel<<<gProj, dim3(256), 0, stream>>>(
        t16, wk, bk + (long)l * DM, k32, (_Float16*)nullptr, ROWS, DM, DM, 0);
    wmma_gemm_kernel<<<gProj, dim3(256), 0, stream>>>(
        t16, wv, bv + (long)l * DM, v32, (_Float16*)nullptr, ROWS, DM, DM, 0);

    attention_kernel<<<dim3(NB * HH), dim3(128), 0, stream>>>(q32, k32, v32, o16);

    wmma_gemm_kernel<<<gProj, dim3(256), 0, stream>>>(
        o16, wo, bo + (long)l * DM, y32, (_Float16*)nullptr, ROWS, DM, DM, 0);

    add_ln_kernel<<<dim3(ROWS), dim3(128), 0, stream>>>(
        t32, y32, g1 + (long)l * DM, be1 + (long)l * DM, t16);

    wmma_gemm_kernel<<<dim3(ROWS / 32, DFFN / 64), dim3(256), 0, stream>>>(
        t16, w1, b1 + (long)l * DFFN, (float*)nullptr, mid16, ROWS, DFFN, DM, 1);
    wmma_gemm_kernel<<<gProj, dim3(256), 0, stream>>>(
        mid16, w2, b2 + (long)l * DM, y32, (_Float16*)nullptr, ROWS, DM, DFFN, 0);

    add_ln_kernel<<<dim3(ROWS), dim3(128), 0, stream>>>(
        t32, y32, g2 + (long)l * DM, be2 + (long)l * DM, t16);
  }

  // 6) classifier head
  classifier_kernel<<<dim3(NB), dim3(256), 0, stream>>>(t32, cls_w, cls_b, out);
}
